// DecoderWithAttention_30597347017123
// MI455X (gfx1250) — matrix operand order repaired and checked by reading, not compiled, not run on hardware
//
#include <hip/hip_runtime.h>
#include <cstddef>

// ---------------- problem constants ----------------
#define BB   64
#define PP   196
#define ENCD 2048
#define ATTD 512
#define EMBD 512
#define HD   512
#define VD   100
#define ML   71
#define TT   70
#define G4   2048   // 4*H
#define KCAT 3072   // EMB+ENC+H

typedef __attribute__((ext_vector_type(16))) __bf16 v16bf;
typedef __attribute__((ext_vector_type(8)))  __bf16 v8bf;
typedef __attribute__((ext_vector_type(8)))  float  v8f;

// =====================================================================
// K0: stable argsort by descending length + gather captions + int->float
//     outputs (caps, dec_lens, sort_ind) written every call.
// =====================================================================
__global__ __launch_bounds__(64) void sort_prep_k(
    const int* __restrict__ lens, const int* __restrict__ caps,
    int* __restrict__ sort_ind, int* __restrict__ dec_lens, int* __restrict__ caps_s,
    float* __restrict__ out_caps, float* __restrict__ out_declens,
    float* __restrict__ out_sortind)
{
    __shared__ int L[BB];
    int i = threadIdx.x;
    L[i] = lens[i];
    __syncthreads();
    int li = L[i];
    int pos = 0;
    for (int j = 0; j < BB; ++j) {
        int lj = L[j];
        if (lj > li || (lj == li && j < i)) ++pos;
    }
    sort_ind[pos]    = i;
    dec_lens[pos]    = li - 1;
    out_declens[pos] = (float)(li - 1);
    out_sortind[pos] = (float)i;
    for (int t = 0; t < ML; ++t) {
        int v = caps[i * ML + t];
        caps_s[pos * ML + t]   = v;
        out_caps[pos * ML + t] = (float)v;
    }
}

// =====================================================================
// K1: gather encoder rows by sort_ind, convert fp32 -> bf16
// =====================================================================
__global__ __launch_bounds__(256) void gather_enc_k(
    const float* __restrict__ enc, const int* __restrict__ sort_ind,
    __bf16* __restrict__ enc_s)
{
    int bp = blockIdx.x;
    int b = bp / PP, p = bp % PP;
    int sb = sort_ind[b];
    const float* src = enc + ((size_t)sb * PP + p) * ENCD;
    __bf16* dst = enc_s + ((size_t)b * PP + p) * ENCD;
    for (int e = threadIdx.x; e < ENCD; e += 256) dst[e] = (__bf16)src[e];
}

// =====================================================================
// K2: mean over P (fp32 source for accuracy), dual fp32/bf16 output
// =====================================================================
__global__ __launch_bounds__(256) void mean_enc_k(
    const float* __restrict__ enc, const int* __restrict__ sort_ind,
    float* __restrict__ mean, __bf16* __restrict__ mean_bf)
{
    int b = blockIdx.y;
    int e = blockIdx.x * 256 + threadIdx.x;
    int sb = sort_ind[b];
    const float* base = enc + (size_t)sb * PP * ENCD + e;
    float s = 0.f;
    for (int p = 0; p < PP; ++p) s += base[(size_t)p * ENCD];
    s *= (1.0f / PP);
    mean[(size_t)b * ENCD + e]    = s;
    mean_bf[(size_t)b * ENCD + e] = (__bf16)s;
}

// =====================================================================
// K3: transpose + convert weights: dst[n*ld + off + k] = src[k*N + n]
// =====================================================================
__global__ __launch_bounds__(256) void transpose_cvt_k(
    const float* __restrict__ src, __bf16* __restrict__ dst,
    int K, int N, int dstLd, int dstOff)
{
    __shared__ float tile[32][33];
    int k0 = blockIdx.y * 32, n0 = blockIdx.x * 32;
    int tx = threadIdx.x & 31, ty = threadIdx.x >> 5;
    for (int i = ty; i < 32; i += 8) {
        int k = k0 + i, n = n0 + tx;
        tile[i][tx] = (k < K && n < N) ? src[(size_t)k * N + n] : 0.f;
    }
    __syncthreads();
    for (int i = ty; i < 32; i += 8) {
        int n = n0 + i, k = k0 + tx;
        if (n < N && k < K)
            dst[(size_t)n * dstLd + dstOff + k] = (__bf16)tile[tx][i];
    }
}

// =====================================================================
// K4: generic WMMA GEMM  C[MxN] = A[MxK](bf16) * Bt[NxK](bf16)^T + bias
//     4 waves/block, each wave: 16 rows x 64 cols, K step 32.
//     M = gridDim.y*64, N = gridDim.x*64, K % 32 == 0.
// =====================================================================
__global__ __launch_bounds__(128) void gemm_bf16_wmma_k(
    const __bf16* __restrict__ A, const __bf16* __restrict__ Bt,
    const float* __restrict__ bias, float* __restrict__ C,
    __bf16* __restrict__ Cbf, int N, int K)
{
    const int lane = threadIdx.x & 31;
    const int wave = threadIdx.x >> 5;
    const int rr   = lane & 15;
    const int hi   = lane >> 4;          // 0 or 1
    const int m0   = blockIdx.y * 64 + wave * 16;
    const int n0   = blockIdx.x * 64;

    v8f acc[4] = {};

    // A fragment: lane<16 holds K 0..15; lane>=16 holds K 16..31
    const __bf16* Ap  = A  + (size_t)(m0 + rr) * K + hi * 16;
    // B fragment (from transposed weight, row n): lane<16 K 0..15, lane>=16 K 16..31
    const __bf16* Bp  = Bt + (size_t)(n0 + rr) * K + hi * 16;
    const size_t  bstr = (size_t)16 * K;

    for (int k0 = 0; k0 < K; k0 += 32) {
        v8bf alo = *(const v8bf*)(Ap + k0);
        v8bf ahi = *(const v8bf*)(Ap + k0 + 8);
        v16bf a = __builtin_shufflevector(alo, ahi,
                    0,1,2,3,4,5,6,7,8,9,10,11,12,13,14,15);
        v16bf b0 = *(const v16bf*)(Bp + k0);
        v16bf b1 = *(const v16bf*)(Bp + bstr + k0);
        v16bf b2 = *(const v16bf*)(Bp + 2 * bstr + k0);
        v16bf b3 = *(const v16bf*)(Bp + 3 * bstr + k0);
        acc[0] = __builtin_amdgcn_wmma_f32_16x16x32_bf16(false, a, false, b0, (short)0, acc[0], false, false);
        acc[1] = __builtin_amdgcn_wmma_f32_16x16x32_bf16(false, a, false, b1, (short)0, acc[1], false, false);
        acc[2] = __builtin_amdgcn_wmma_f32_16x16x32_bf16(false, a, false, b2, (short)0, acc[2], false, false);
        acc[3] = __builtin_amdgcn_wmma_f32_16x16x32_bf16(false, a, false, b3, (short)0, acc[3], false, false);
    }

    // C/D layout: lane<16 -> col=lane, vgpr i -> row i; lane>=16 -> row i+8
    #pragma unroll
    for (int n = 0; n < 4; ++n) {
        int col = n0 + n * 16 + rr;
        float bv = bias ? bias[col] : 0.0f;
        #pragma unroll
        for (int i = 0; i < 8; ++i) {
            int row = m0 + hi * 8 + i;
            float v = acc[n][i] + bv;
            C[(size_t)row * N + col] = v;
            if (Cbf) Cbf[(size_t)row * N + col] = (__bf16)v;
        }
    }
}

// =====================================================================
// K5: attention scores: dot(tanh(att_enc + dec_att), W_full) + b_full
// =====================================================================
__global__ __launch_bounds__(256) void scores_k(
    const float* __restrict__ att_enc, const float* __restrict__ dec_att,
    const float* __restrict__ W_full, const float* __restrict__ b_full,
    float* __restrict__ scores)
{
    int p = blockIdx.x, b = blockIdx.y, tid = threadIdx.x;
    const float* ae = att_enc + ((size_t)b * PP + p) * ATTD;
    const float* da = dec_att + (size_t)b * ATTD;
    float s = 0.f;
    for (int a = tid; a < ATTD; a += 256) s += tanhf(ae[a] + da[a]) * W_full[a];
    __shared__ float red[256];
    red[tid] = s; __syncthreads();
    for (int w = 128; w > 0; w >>= 1) { if (tid < w) red[tid] += red[tid + w]; __syncthreads(); }
    if (tid == 0) scores[(size_t)b * PP + p] = red[0] + b_full[0];
}

// =====================================================================
// K6: softmax over P; writes alpha (unmasked, for awe) + masked out_alphas
// =====================================================================
__global__ __launch_bounds__(256) void softmax_k(
    const float* __restrict__ scores, const int* __restrict__ dec_lens,
    float* __restrict__ alpha, float* __restrict__ out_alphas, int t)
{
    int b = blockIdx.x, tid = threadIdx.x;
    __shared__ float red[256];
    float v = (tid < PP) ? scores[(size_t)b * PP + tid] : -3.4e38f;
    red[tid] = v; __syncthreads();
    for (int w = 128; w > 0; w >>= 1) { if (tid < w) red[tid] = fmaxf(red[tid], red[tid + w]); __syncthreads(); }
    float m = red[0]; __syncthreads();
    float e = (tid < PP) ? expf(v - m) : 0.f;
    red[tid] = e; __syncthreads();
    for (int w = 128; w > 0; w >>= 1) { if (tid < w) red[tid] += red[tid + w]; __syncthreads(); }
    float s = red[0];
    if (tid < PP) {
        float a = e / s;
        alpha[(size_t)b * PP + tid] = a;
        float msk = (t < dec_lens[b]) ? 1.f : 0.f;
        out_alphas[((size_t)b * TT + t) * PP + tid] = a * msk;
    }
}

// =====================================================================
// K7: attention-weighted encoding: awe[b,e] = sum_p alpha[b,p]*enc_s[b,p,e]
// =====================================================================
__global__ __launch_bounds__(256) void awe_k(
    const __bf16* __restrict__ enc_s, const float* __restrict__ alpha,
    float* __restrict__ awe)
{
    int b = blockIdx.y;
    int e = blockIdx.x * 256 + threadIdx.x;
    const __bf16* base = enc_s + (size_t)b * PP * ENCD + e;
    const float*  al   = alpha + (size_t)b * PP;
    float s = 0.f;
    for (int p = 0; p < PP; ++p) s += al[p] * (float)base[(size_t)p * ENCD];
    awe[(size_t)b * ENCD + e] = s;
}

// =====================================================================
// K8: build A2 = [emb(t) | sigmoid(beta_lin)*awe | h] as bf16 (64 x 3072)
// =====================================================================
__global__ __launch_bounds__(256) void build_x_k(
    const float* __restrict__ W_embed, const int* __restrict__ caps_s,
    const float* __restrict__ beta_lin, const float* __restrict__ awe,
    const __bf16* __restrict__ h_bf, __bf16* __restrict__ A2, int t)
{
    int b = blockIdx.x, tid = threadIdx.x;
    __bf16* row = A2 + (size_t)b * KCAT;
    int tok = caps_s[b * ML + t];
    const float* erow = W_embed + (size_t)tok * EMBD;
    for (int j = tid; j < EMBD; j += 256) row[j] = (__bf16)erow[j];
    for (int j = tid; j < ENCD; j += 256) {
        float g = 1.f / (1.f + expf(-beta_lin[(size_t)b * ENCD + j]));
        row[EMBD + j] = (__bf16)(g * awe[(size_t)b * ENCD + j]);
    }
    for (int j = tid; j < HD; j += 256) row[EMBD + ENCD + j] = h_bf[(size_t)b * HD + j];
}

// =====================================================================
// K9: LSTM pointwise update (i,f,g,o), h -> fp32 + bf16
// =====================================================================
__global__ __launch_bounds__(256) void lstm_k(
    const float* __restrict__ gates, const float* __restrict__ b_ih,
    const float* __restrict__ b_hh, float* __restrict__ c,
    float* __restrict__ h, __bf16* __restrict__ h_bf)
{
    int b = blockIdx.x;
    for (int j = threadIdx.x; j < HD; j += 256) {
        const float* gr = gates + (size_t)b * G4;
        float gi = gr[j]            + b_ih[j]            + b_hh[j];
        float gf = gr[HD + j]       + b_ih[HD + j]       + b_hh[HD + j];
        float gg = gr[2 * HD + j]   + b_ih[2 * HD + j]   + b_hh[2 * HD + j];
        float go = gr[3 * HD + j]   + b_ih[3 * HD + j]   + b_hh[3 * HD + j];
        float i_ = 1.f / (1.f + expf(-gi));
        float f_ = 1.f / (1.f + expf(-gf));
        float g_ = tanhf(gg);
        float o_ = 1.f / (1.f + expf(-go));
        float cc = f_ * c[(size_t)b * HD + j] + i_ * g_;
        c[(size_t)b * HD + j] = cc;
        float hh = o_ * tanhf(cc);
        h[(size_t)b * HD + j]    = hh;
        h_bf[(size_t)b * HD + j] = (__bf16)hh;
    }
}

// =====================================================================
// K10: preds = mask * (h @ W_fc + b_fc)   (V=100, tiny -> LDS staged VALU)
// =====================================================================
__global__ __launch_bounds__(128) void preds_k(
    const float* __restrict__ h, const float* __restrict__ W_fc,
    const float* __restrict__ b_fc, const int* __restrict__ dec_lens,
    float* __restrict__ out_preds, int t)
{
    int b = blockIdx.x;
    __shared__ float hs[HD];
    for (int j = threadIdx.x; j < HD; j += 128) hs[j] = h[(size_t)b * HD + j];
    __syncthreads();
    int v = threadIdx.x;
    if (v < VD) {
        float s = b_fc[v];
        for (int a = 0; a < HD; ++a) s += hs[a] * W_fc[a * VD + v];
        float msk = (t < dec_lens[b]) ? 1.f : 0.f;
        out_preds[((size_t)b * TT + t) * VD + v] = s * msk;
    }
}

// =====================================================================
// host orchestration
// =====================================================================
extern "C" void kernel_launch(void* const* d_in, const int* in_sizes, int n_in,
                              void* d_out, int out_size, void* d_ws, size_t ws_size,
                              hipStream_t stream)
{
    (void)in_sizes; (void)n_in; (void)out_size; (void)ws_size;
    const float* enc       = (const float*)d_in[0];
    const int*   caps      = (const int*)  d_in[1];
    const int*   lens      = (const int*)  d_in[2];
    const float* W_embed   = (const float*)d_in[3];
    const float* W_enc_att = (const float*)d_in[4];
    const float* b_enc_att = (const float*)d_in[5];
    const float* W_dec_att = (const float*)d_in[6];
    const float* b_dec_att = (const float*)d_in[7];
    const float* W_full    = (const float*)d_in[8];
    const float* b_full    = (const float*)d_in[9];
    const float* W_init_h  = (const float*)d_in[10];
    const float* b_init_h  = (const float*)d_in[11];
    const float* W_init_c  = (const float*)d_in[12];
    const float* b_init_c  = (const float*)d_in[13];
    const float* W_beta    = (const float*)d_in[14];
    const float* b_beta    = (const float*)d_in[15];
    const float* W_ih      = (const float*)d_in[16];
    const float* b_ih      = (const float*)d_in[17];
    const float* W_hh      = (const float*)d_in[18];
    const float* b_hh      = (const float*)d_in[19];
    const float* W_fc      = (const float*)d_in[20];
    const float* b_fc      = (const float*)d_in[21];

    // ---- output layout (floats, concatenated in return order) ----
    float* out          = (float*)d_out;
    float* out_preds    = out;                               // 64*70*100
    float* out_caps     = out + (size_t)BB * TT * VD;        // 64*71
    float* out_declens  = out_caps + (size_t)BB * ML;        // 64
    float* out_alphas   = out_declens + BB;                  // 64*70*196
    float* out_sortind  = out_alphas + (size_t)BB * TT * PP; // 64

    // ---- workspace carve-up ----
    char* ws = (char*)d_ws;
    size_t off = 0;
    auto alloc = [&](size_t bytes) -> char* {
        char* p = ws + off;
        off = (off + bytes + 255) & ~(size_t)255;
        return p;
    };
    __bf16* enc_s     = (__bf16*)alloc((size_t)BB * PP * ENCD * 2);
    float*  att_enc   = (float*) alloc((size_t)BB * PP * ATTD * 4);
    __bf16* Wt_enc    = (__bf16*)alloc((size_t)ATTD * ENCD * 2);   // 512 x 2048
    __bf16* Wt_init_h = (__bf16*)alloc((size_t)HD * ENCD * 2);     // 512 x 2048
    __bf16* Wt_init_c = (__bf16*)alloc((size_t)HD * ENCD * 2);
    __bf16* Wt_dec    = (__bf16*)alloc((size_t)ATTD * HD * 2);     // 512 x 512
    __bf16* Wt_beta   = (__bf16*)alloc((size_t)ENCD * HD * 2);     // 2048 x 512
    __bf16* Wcat_t    = (__bf16*)alloc((size_t)G4 * KCAT * 2);     // 2048 x 3072
    float*  mean      = (float*) alloc((size_t)BB * ENCD * 4);
    __bf16* mean_bf   = (__bf16*)alloc((size_t)BB * ENCD * 2);
    float*  h         = (float*) alloc((size_t)BB * HD * 4);
    float*  c         = (float*) alloc((size_t)BB * HD * 4);
    __bf16* h_bf      = (__bf16*)alloc((size_t)BB * HD * 2);
    float*  dec_att   = (float*) alloc((size_t)BB * ATTD * 4);
    float*  beta_lin  = (float*) alloc((size_t)BB * ENCD * 4);
    float*  awe       = (float*) alloc((size_t)BB * ENCD * 4);
    __bf16* A2        = (__bf16*)alloc((size_t)BB * KCAT * 2);
    float*  gates     = (float*) alloc((size_t)BB * G4 * 4);
    float*  scores    = (float*) alloc((size_t)BB * PP * 4);
    float*  alpha     = (float*) alloc((size_t)BB * PP * 4);
    int*    sort_ind  = (int*)   alloc(BB * 4);
    int*    dec_lens  = (int*)   alloc(BB * 4);
    int*    caps_s    = (int*)   alloc((size_t)BB * ML * 4);

    // ---- setup ----
    sort_prep_k<<<1, 64, 0, stream>>>(lens, caps, sort_ind, dec_lens, caps_s,
                                      out_caps, out_declens, out_sortind);
    gather_enc_k<<<BB * PP, 256, 0, stream>>>(enc, sort_ind, enc_s);
    mean_enc_k<<<dim3(ENCD / 256, BB), 256, 0, stream>>>(enc, sort_ind, mean, mean_bf);

    // weight transposes (fp32 KxN -> bf16 NxK)
    transpose_cvt_k<<<dim3(ATTD / 32, ENCD / 32), 256, 0, stream>>>(W_enc_att, Wt_enc,  ENCD, ATTD, ENCD, 0);
    transpose_cvt_k<<<dim3(HD / 32,  ENCD / 32), 256, 0, stream>>>(W_init_h,  Wt_init_h, ENCD, HD,  ENCD, 0);
    transpose_cvt_k<<<dim3(HD / 32,  ENCD / 32), 256, 0, stream>>>(W_init_c,  Wt_init_c, ENCD, HD,  ENCD, 0);
    transpose_cvt_k<<<dim3(ATTD / 32, HD / 32),  256, 0, stream>>>(W_dec_att, Wt_dec,    HD,  ATTD, HD,   0);
    transpose_cvt_k<<<dim3(ENCD / 32, HD / 32),  256, 0, stream>>>(W_beta,    Wt_beta,   HD,  ENCD, HD,   0);
    transpose_cvt_k<<<dim3(G4 / 32, (EMBD + ENCD) / 32), 256, 0, stream>>>(W_ih, Wcat_t, EMBD + ENCD, G4, KCAT, 0);
    transpose_cvt_k<<<dim3(G4 / 32, HD / 32),    256, 0, stream>>>(W_hh,      Wcat_t,    HD,  G4,  KCAT, EMBD + ENCD);

    // h0 / c0 = mean_enc @ W_init_{h,c} + b  (WMMA, M=64, N=512, K=2048)
    gemm_bf16_wmma_k<<<dim3(HD / 64, 1), 128, 0, stream>>>(mean_bf, Wt_init_h, b_init_h, h, h_bf, HD, ENCD);
    gemm_bf16_wmma_k<<<dim3(HD / 64, 1), 128, 0, stream>>>(mean_bf, Wt_init_c, b_init_c, c, nullptr, HD, ENCD);

    // att_enc = enc_s @ W_enc_att + b  (WMMA, M=12544, N=512, K=2048)
    gemm_bf16_wmma_k<<<dim3(ATTD / 64, (BB * PP) / 64), 128, 0, stream>>>(
        enc_s, Wt_enc, b_enc_att, att_enc, nullptr, ATTD, ENCD);

    // ---- decode loop ----
    for (int t = 0; t < TT; ++t) {
        // dec_att = h @ W_dec_att + b  (64x512x512)
        gemm_bf16_wmma_k<<<dim3(ATTD / 64, 1), 128, 0, stream>>>(
            h_bf, Wt_dec, b_dec_att, dec_att, nullptr, ATTD, HD);
        // attention scores + softmax + weighted encoding
        scores_k<<<dim3(PP, BB), 256, 0, stream>>>(att_enc, dec_att, W_full, b_full, scores);
        softmax_k<<<BB, 256, 0, stream>>>(scores, dec_lens, alpha, out_alphas, t);
        awe_k<<<dim3(ENCD / 256, BB), 256, 0, stream>>>(enc_s, alpha, awe);
        // gate pre-activation: beta_lin = h @ W_beta + b  (64x512x2048)
        gemm_bf16_wmma_k<<<dim3(ENCD / 64, 1), 128, 0, stream>>>(
            h_bf, Wt_beta, b_beta, beta_lin, nullptr, ENCD, HD);
        // A2 = [emb_t | sigmoid(beta_lin)*awe | h]
        build_x_k<<<BB, 256, 0, stream>>>(W_embed, caps_s, beta_lin, awe, h_bf, A2, t);
        // gates = A2 @ [W_ih; W_hh]  (64x3072x2048, biases folded into lstm_k)
        gemm_bf16_wmma_k<<<dim3(G4 / 64, 1), 128, 0, stream>>>(
            A2, Wcat_t, nullptr, gates, nullptr, G4, KCAT);
        // LSTM pointwise + output head
        lstm_k<<<BB, 256, 0, stream>>>(gates, b_ih, b_hh, c, h, h_bf);
        preds_k<<<BB, 128, 0, stream>>>(h, W_fc, b_fc, dec_lens, out_preds, t);
    }
}
